// DNGPU_66932770341352
// MI455X (gfx1250) — matrix operand order repaired and checked by reading, not compiled, not run on hardware
//
#include <hip/hip_runtime.h>

// ---------------------------------------------------------------------------
// DNGPU cell on MI455X (gfx1250):
//  - 32 workgroups (one per batch element), 768 threads = 24 wave32.
//  - Wave = (mhalf 0..1, ntile 0..11): 4 mTiles x 1 nTile of 16x16 output.
//  - State [128 x 192] in LDS as bf16, 1 zero pad row top+bottom (SAME conv
//    padding + shift layer for free). Double-buffered; 128 steps, no global
//    round-trips inside the loop.
//  - (tap,kc) loop kept ROLLED (pragma unroll 1) with incrementing pointers:
//    since 192 = 6*32, the A offset advances +32 elems/iter and the B frag
//    pointer by a constant stride. Bounded live set => no scratch spills,
//    and LICM cannot hoist the weight loads (loop-carried pointers).
//  - gate stashed as bf16 into the next-state buffer slot this same thread
//    later overwrites with out -> zero extra registers/LDS for gate.
//  - sigmoid/tanh use v_exp_f32 + v_rcp_f32 (no precise-division sequences).
// ---------------------------------------------------------------------------

typedef __attribute__((ext_vector_type(16))) __bf16 v16bf;
typedef __attribute__((ext_vector_type(8)))  __bf16 bf16x8;
typedef __attribute__((ext_vector_type(8)))  float  v8f;

#define BB   32
#define LL   128
#define CC   192
#define LPAD 130                 // 128 rows + 1 zero row top + bottom
#define ELEMS (LPAD * CC)        // 24960 bf16 per buffer
#define NTILES 12                // 12 column tiles of 16 channels
#define NWAVES 24                // 2 m-halves x 12 nTiles
#define NTHREADS (NWAVES * 32)   // 768
#define MT_PER 4                 // mTiles per wave (128/16/2)
#define KCHUNKS 6                // 192 / 32
#define FRAG_ELEMS 512           // 32x16 bf16 B fragment
#define TKITERS (3 * KCHUNKS)    // flattened tap*kc loop = 18
#define FRAG_STRIDE (NTILES * FRAG_ELEMS)        // per (tap,kc) step = 6144
#define CONV_STRIDE (3 * KCHUNKS * NTILES * FRAG_ELEMS)  // per conv = 110592
#define NFRAGS (3 * 3 * KCHUNKS * NTILES)        // 648

// Fast sigmoid/tanh: one v_exp_f32 + one v_rcp_f32 each (bf16-state accuracy).
__device__ __forceinline__ float fsigmoid(float x) {
    return __builtin_amdgcn_rcpf(1.0f + __expf(-x));
}
__device__ __forceinline__ float ftanh(float x) {
    return 1.0f - 2.0f * __builtin_amdgcn_rcpf(__expf(2.0f * x) + 1.0f);
}

// ---------------------------------------------------------------------------
// Prep: repack w[3][192][192] fp32 (per conv) into bf16 WMMA B fragments.
// Fragment (conv,tap,kc,ntile): lane n, vgpr v, half h holds
//   W[tap][ ci = kc*32 + (n>=16?16:0) + 2v + h ][ co = ntile*16 + (n&15) ]
// stored so each lane's 16 bf16 are contiguous (2 x b128 loads per lane).
// ---------------------------------------------------------------------------
__global__ void __launch_bounds__(256) dngpu_prep_weights(
    const float* __restrict__ wr, const float* __restrict__ wg,
    const float* __restrict__ wc, __bf16* __restrict__ ws)
{
    int t = blockIdx.x * blockDim.x + threadIdx.x;
    const int total = NFRAGS * FRAG_ELEMS;
    if (t >= total) return;
    int frag = t >> 9;           // /512
    int r    = t & 511;
    int lane = r >> 4;           // 0..31
    int idx  = r & 15;           // element within lane
    int v    = idx >> 1;
    int h    = idx & 1;
    int ntile = frag % NTILES;
    int kc    = (frag / NTILES) % KCHUNKS;
    int tap   = (frag / (NTILES * KCHUNKS)) % 3;
    int conv  = frag / (NTILES * KCHUNKS * 3);
    int ci = kc * 32 + ((lane >= 16) ? 16 : 0) + 2 * v + h;
    int co = ntile * 16 + (lane & 15);
    const float* w = (conv == 0) ? wr : (conv == 1) ? wg : wc;
    float val = w[(tap * CC + ci) * CC + co];
    ws[frag * FRAG_ELEMS + lane * 16 + idx] = (__bf16)val;
}

__device__ __forceinline__ v16bf load_frag16(const __bf16* p, int ofsElems) {
    v16bf f;
    ((bf16x8*)&f)[0] = *(const bf16x8*)(p + ofsElems);
    ((bf16x8*)&f)[1] = *(const bf16x8*)(p + ofsElems + 8);
    return f;
}

// A fragment: 16 contiguous K elements per lane (K 16*hi .. 16*hi+15).
__device__ __forceinline__ v16bf load_afrag16(const __bf16* p, int ofsElems) {
    v16bf a;
    ((bf16x8*)&a)[0] = *(const bf16x8*)(p + ofsElems);
    ((bf16x8*)&a)[1] = *(const bf16x8*)(p + ofsElems + 8);
    return a;
}

__global__ void __launch_bounds__(NTHREADS) dngpu_cell_kernel(
    const float* __restrict__ x,
    const float* __restrict__ br, const float* __restrict__ bg,
    const float* __restrict__ bc,
    const __bf16* __restrict__ ws,
    float* __restrict__ out)
{
    extern __shared__ __bf16 smem[];
    __bf16* buf0 = smem;               // state buffer A (padded)
    __bf16* buf1 = smem + ELEMS;       // state buffer B (padded)
    __bf16* rmem = smem + 2 * ELEMS;   // reset * mem   (padded)

    const int b     = blockIdx.x;
    const int tid   = threadIdx.x;
    const int wv    = tid >> 5;
    const int ntile = wv % NTILES;     // output channel tile
    const int mbase = (wv / NTILES) * MT_PER;  // first mTile (0 or 4)
    const int lane  = tid & 31;
    const int laneM = lane & 15;
    const int hi    = lane >> 4;       // 0 | 1

    // ---- init: zero (pad rows!) then load x into rows 1..128 of buf0 ----
    for (int i = tid; i < ELEMS; i += NTHREADS) {
        buf0[i] = (__bf16)0.0f;
        buf1[i] = (__bf16)0.0f;
        rmem[i] = (__bf16)0.0f;
    }
    __syncthreads();
    const float* xb = x + (size_t)b * LL * CC;
    for (int i = tid; i < LL * CC; i += NTHREADS)
        buf0[i + CC] = (__bf16)xb[i];          // (l,c) -> row l+1
    __syncthreads();

    const int col = ntile * 16 + laneM;        // this lane's output channel
    const float biasR = br[col];
    const float biasG = bg[col];
    const float biasC = bc[col];

    // Per-lane constant offsets.
    //   A base (elems, padded row laneM, tap=0,kc=0): (mbase*16 + laneM)*CC + hi*16
    //   advances +32 elems per flattened (tap,kc) iteration (CC == 6*32).
    const int aBaseOfs = (mbase * 16 + laneM) * CC + hi * 16;
    const __bf16* wBase = ws + ntile * FRAG_ELEMS + lane * 16;

    __bf16* cur = buf0;
    __bf16* nxt = buf1;

    for (int step = 0; step < LL; ++step) {
        // ========= phase A: reset & gate convs (share A fragments) =========
        v8f accR[MT_PER], accG[MT_PER];
        v8f vzero = {};
        #pragma unroll
        for (int m = 0; m < MT_PER; ++m) { accR[m] = vzero; accG[m] = vzero; }

        {
            const __bf16* ap = cur + aBaseOfs;
            const __bf16* wp = wBase;
            #pragma unroll 1
            for (int t = 0; t < TKITERS; ++t) {
                v16bf Br = load_frag16(wp, 0);
                v16bf Bg = load_frag16(wp, CONV_STRIDE);
                #pragma unroll
                for (int m = 0; m < MT_PER; ++m) {
                    v16bf A = load_afrag16(ap, m * 16 * CC);
                    accR[m] = __builtin_amdgcn_wmma_f32_16x16x32_bf16(
                        false, A, false, Br, (short)0, accR[m], false, false);
                    accG[m] = __builtin_amdgcn_wmma_f32_16x16x32_bf16(
                        false, A, false, Bg, (short)0, accG[m], false, false);
                }
                ap += 32;
                wp += FRAG_STRIDE;
            }
        }

        // sigmoid(reset) -> rmem = reset*mem; sigmoid(gate) stashed into nxt
        // (same thread re-reads and overwrites that slot in phase B epilogue).
        #pragma unroll
        for (int m = 0; m < MT_PER; ++m) {
            const int rowBase = (mbase + m) * 16 + hi * 8;
            #pragma unroll
            for (int r = 0; r < 8; ++r) {
                const int l = rowBase + r;
                const int idx = (l + 1) * CC + col;
                float reset = fsigmoid(accR[m][r] + biasR);
                float gate  = fsigmoid(accG[m][r] + biasG);
                nxt[idx]  = (__bf16)gate;                       // stash gate
                rmem[idx] = (__bf16)(reset * (float)cur[idx]);
            }
        }
        __syncthreads();   // rmem complete before cand conv reads it

        // ================= phase B: cand conv from rmem ====================
        v8f accC[MT_PER];
        #pragma unroll
        for (int m = 0; m < MT_PER; ++m) accC[m] = vzero;

        {
            const __bf16* ap = rmem + aBaseOfs;
            const __bf16* wp = wBase + 2 * CONV_STRIDE;
            #pragma unroll 1
            for (int t = 0; t < TKITERS; ++t) {
                v16bf Bc = load_frag16(wp, 0);
                #pragma unroll
                for (int m = 0; m < MT_PER; ++m) {
                    v16bf A = load_afrag16(ap, m * 16 * CC);
                    accC[m] = __builtin_amdgcn_wmma_f32_16x16x32_bf16(
                        false, A, false, Bc, (short)0, accC[m], false, false);
                }
                ap += 32;
                wp += FRAG_STRIDE;
            }
        }

        // out = gate * shift(mem) + (1-gate) * tanh(cand); write to nxt.
        #pragma unroll
        for (int m = 0; m < MT_PER; ++m) {
            const int rowBase = (mbase + m) * 16 + hi * 8;
            #pragma unroll
            for (int r = 0; r < 8; ++r) {
                const int l = rowBase + r;
                const int idx = (l + 1) * CC + col;
                float cand = ftanh(accC[m][r] + biasC);
                float gate = (float)nxt[idx];                  // un-stash gate
                // shifted[l] = mem[l-1]; padded index (l-1)+1 = l (row 0 = 0)
                float shifted = (float)cur[l * CC + col];
                nxt[idx] = (__bf16)(gate * shifted + (1.0f - gate) * cand);
            }
        }
        __syncthreads();   // nxt complete before next step reads it

        __bf16* t = cur; cur = nxt; nxt = t;
    }

    // ---- write final state fp32 ----
    float* ob = out + (size_t)b * LL * CC;
    for (int i = tid; i < LL * CC; i += NTHREADS)
        ob[i] = (float)cur[i + CC];
}

extern "C" void kernel_launch(void* const* d_in, const int* in_sizes, int n_in,
                              void* d_out, int out_size, void* d_ws, size_t ws_size,
                              hipStream_t stream) {
    (void)in_sizes; (void)n_in; (void)out_size; (void)ws_size;
    const float* x  = (const float*)d_in[0];
    const float* wr = (const float*)d_in[1];
    const float* br = (const float*)d_in[2];
    const float* wg = (const float*)d_in[3];
    const float* bg = (const float*)d_in[4];
    const float* wc = (const float*)d_in[5];
    const float* bc = (const float*)d_in[6];
    __bf16* ws = (__bf16*)d_ws;         // 648 frags * 1 KB = 663552 B
    float* out = (float*)d_out;

    const int prepTotal = NFRAGS * FRAG_ELEMS;      // 331776
    dngpu_prep_weights<<<(prepTotal + 255) / 256, 256, 0, stream>>>(wr, wg, wc, ws);

    const size_t shmem = (size_t)3 * ELEMS * sizeof(__bf16);   // 149760 B
    dngpu_cell_kernel<<<BB, NTHREADS, shmem, stream>>>(x, br, bg, bc, ws, out);
}
